// JointNetwork_24618752541227
// MI455X (gfx1250) — matrix operand order repaired and checked by reading, not compiled, not run on hardware
//
#include <hip/hip_runtime.h>

typedef __attribute__((ext_vector_type(16))) _Float16 v16h;
typedef __attribute__((ext_vector_type(8)))  _Float16 v8h;
typedef __attribute__((ext_vector_type(8)))  float    v8f;
typedef __attribute__((ext_vector_type(4)))  float    v4f;

#define DIM_   256
#define VOCAB_ 1024
#define T_     256
#define U_     64
#define B_     4

#define XN_ (B_ * T_ * DIM_)     // 262144
#define YN_ (B_ * U_ * DIM_)     // 65536
#define WN_ (VOCAB_ * DIM_)      // 262144

// ws layout (f16): [Wh: WN_][xh: XN_][yh: YN_]
#define WH_OFF_ 0
#define XH_OFF_ ((size_t)WN_)
#define YH_OFF_ ((size_t)(WN_ + XN_))
#define WS_NEED_ ((size_t)(WN_ + XN_ + YN_) * sizeof(_Float16))

// ---------------------------------------------------------------------------
// Pre-pass: f32 -> f16 downcast of W, x, y into workspace (one-time, ~1.2 MB).
// ---------------------------------------------------------------------------
__global__ __launch_bounds__(256)
void cvt_f16_kernel(const float* __restrict__ x, const float* __restrict__ y,
                    const float* __restrict__ W,
                    _Float16* __restrict__ xh, _Float16* __restrict__ yh,
                    _Float16* __restrict__ Wh)
{
    const int i = (blockIdx.x * 256 + threadIdx.x) * 8;
    if (i < WN_) {
        const v4f a0 = *(const v4f*)(W + i);
        const v4f a1 = *(const v4f*)(W + i + 4);
        const v4f b0 = *(const v4f*)(x + i);
        const v4f b1 = *(const v4f*)(x + i + 4);
        v8h ow, ox;
#pragma unroll
        for (int j = 0; j < 4; ++j) {
            ow[j] = (_Float16)a0[j]; ow[4 + j] = (_Float16)a1[j];
            ox[j] = (_Float16)b0[j]; ox[4 + j] = (_Float16)b1[j];
        }
        *(v8h*)(Wh + i) = ow;
        *(v8h*)(xh + i) = ox;
    }
    if (i < YN_) {
        const v4f c0 = *(const v4f*)(y + i);
        const v4f c1 = *(const v4f*)(y + i + 4);
        v8h oy;
#pragma unroll
        for (int j = 0; j < 4; ++j) { oy[j] = (_Float16)c0[j]; oy[4 + j] = (_Float16)c1[j]; }
        *(v8h*)(yh + i) = oy;
    }
}

// ---------------------------------------------------------------------------
// Fast path: f16 operands pre-staged. Grid (VOCAB/256, T, B), 256 thr = 8 waves.
// Wave w: u-tile (w&3)*16, v-strip blockIdx.x*256 + (w>>2)*128 (8 tiles of 16).
// ---------------------------------------------------------------------------
__global__ __launch_bounds__(256)
void joint_wmma_f16_kernel(const _Float16* __restrict__ xh,
                           const _Float16* __restrict__ yh,
                           const _Float16* __restrict__ Wh,
                           const float* __restrict__ bias,
                           float* __restrict__ out)
{
    const int lane = threadIdx.x & 31;
    const int wave = threadIdx.x >> 5;
    const int t    = blockIdx.y;
    const int b    = blockIdx.z;

    const int u0     = (wave & 3) * 16;
    const int v_base = blockIdx.x * 256 + (wave >> 2) * 128;

    const int  nlo = lane & 15;
    const bool hi  = (lane >= 16);

    const _Float16* xrow = xh + (size_t)(b * T_ + t) * DIM_;
    const _Float16* yrow = yh + (size_t)(b * U_ + u0 + nlo) * DIM_;

    v8f acc[8];
#pragma unroll
    for (int n = 0; n < 8; ++n) {
        const float bv = bias[v_base + n * 16 + nlo];
#pragma unroll
        for (int j = 0; j < 8; ++j) acc[n][j] = bv;
    }

#pragma unroll 1
    for (int d0 = 0; d0 < DIM_; d0 += 32) {
        // A fragment (16-bit 16x32 layout): lane<16 K=d0+{0..7,16..23},
        // lane>=16 K=d0+{8..15,24..31}. Products via v_pk_mul_f16.
        const int ka = d0 + (hi ? 8 : 0);
        const v8h xa0 = *(const v8h*)(xrow + ka);
        const v8h xa1 = *(const v8h*)(xrow + ka + 16);
        const v8h ya0 = *(const v8h*)(yrow + ka);
        const v8h ya1 = *(const v8h*)(yrow + ka + 16);
        const v8h p0 = xa0 * ya0;
        const v8h p1 = xa1 * ya1;
        v16h a;
#pragma unroll
        for (int i = 0; i < 8; ++i) { a[i] = p0[i]; a[8 + i] = p1[i]; }

        const int kw = d0 + (hi ? 8 : 0);
#pragma unroll
        for (int n = 0; n < 8; ++n) {
            const _Float16* wrow = Wh + (size_t)(v_base + n * 16 + nlo) * DIM_ + kw;
            const v8h b0 = *(const v8h*)(wrow);
            const v8h b1 = *(const v8h*)(wrow + 16);
            v16h bf;
#pragma unroll
            for (int i = 0; i < 8; ++i) { bf[i] = b0[i]; bf[8 + i] = b1[i]; }

            acc[n] = __builtin_amdgcn_wmma_f32_16x16x32_f16(
                false, a, false, bf, (short)0, acc[n], false, false);
        }
    }

    const size_t row_base = ((size_t)(b * T_ + t) * U_ + u0) * VOCAB_;
#pragma unroll
    for (int n = 0; n < 8; ++n) {
        const int v = v_base + n * 16 + nlo;
#pragma unroll
        for (int r = 0; r < 8; ++r) {
            const int urow = r + (hi ? 8 : 0);
            out[row_base + (size_t)urow * VOCAB_ + v] = acc[n][r];
        }
    }
}

// ---------------------------------------------------------------------------
// Fallback (ws too small): round-1 kernel, f32 loads + inline cvt.
// ---------------------------------------------------------------------------
__global__ __launch_bounds__(256)
void joint_wmma_f32src_kernel(const float* __restrict__ x,
                              const float* __restrict__ y,
                              const float* __restrict__ W,
                              const float* __restrict__ bias,
                              float* __restrict__ out)
{
    const int lane = threadIdx.x & 31;
    const int wave = threadIdx.x >> 5;
    const int t    = blockIdx.y;
    const int b    = blockIdx.z;

    const int u0     = (wave & 3) * 16;
    const int v_base = blockIdx.x * 256 + (wave >> 2) * 128;

    const int  nlo = lane & 15;
    const bool hi  = (lane >= 16);

    const float* xrow = x + (size_t)(b * T_ + t) * DIM_;
    const float* yrow = y + (size_t)(b * U_ + u0 + nlo) * DIM_;

    v8f acc[8];
#pragma unroll
    for (int n = 0; n < 8; ++n) {
        const float bv = bias[v_base + n * 16 + nlo];
#pragma unroll
        for (int j = 0; j < 8; ++j) acc[n][j] = bv;
    }

#pragma unroll 1
    for (int d0 = 0; d0 < DIM_; d0 += 32) {
        const int ka = d0 + (hi ? 8 : 0);
        const v4f x0 = *(const v4f*)(xrow + ka);
        const v4f x1 = *(const v4f*)(xrow + ka + 4);
        const v4f x2 = *(const v4f*)(xrow + ka + 16);
        const v4f x3 = *(const v4f*)(xrow + ka + 20);
        const v4f y0 = *(const v4f*)(yrow + ka);
        const v4f y1 = *(const v4f*)(yrow + ka + 4);
        const v4f y2 = *(const v4f*)(yrow + ka + 16);
        const v4f y3 = *(const v4f*)(yrow + ka + 20);

        v16h a;
#pragma unroll
        for (int i = 0; i < 4; ++i) {
            a[i]      = (_Float16)(x0[i] * y0[i]);
            a[4 + i]  = (_Float16)(x1[i] * y1[i]);
            a[8 + i]  = (_Float16)(x2[i] * y2[i]);
            a[12 + i] = (_Float16)(x3[i] * y3[i]);
        }

        const int kw = d0 + (hi ? 8 : 0);
#pragma unroll
        for (int n = 0; n < 8; ++n) {
            const float* wrow = W + (size_t)(v_base + n * 16 + nlo) * DIM_ + kw;
            const v4f w0 = *(const v4f*)(wrow);
            const v4f w1 = *(const v4f*)(wrow + 4);
            const v4f w2 = *(const v4f*)(wrow + 16);
            const v4f w3 = *(const v4f*)(wrow + 20);
            v16h bf;
#pragma unroll
            for (int i = 0; i < 4; ++i) {
                bf[i]      = (_Float16)w0[i];
                bf[4 + i]  = (_Float16)w1[i];
                bf[8 + i]  = (_Float16)w2[i];
                bf[12 + i] = (_Float16)w3[i];
            }
            acc[n] = __builtin_amdgcn_wmma_f32_16x16x32_f16(
                false, a, false, bf, (short)0, acc[n], false, false);
        }
    }

    const size_t row_base = ((size_t)(b * T_ + t) * U_ + u0) * VOCAB_;
#pragma unroll
    for (int n = 0; n < 8; ++n) {
        const int v = v_base + n * 16 + nlo;
#pragma unroll
        for (int r = 0; r < 8; ++r) {
            const int urow = r + (hi ? 8 : 0);
            out[row_base + (size_t)urow * VOCAB_ + v] = acc[n][r];
        }
    }
}

extern "C" void kernel_launch(void* const* d_in, const int* in_sizes, int n_in,
                              void* d_out, int out_size, void* d_ws, size_t ws_size,
                              hipStream_t stream) {
    (void)in_sizes; (void)n_in; (void)out_size;
    const float* x    = (const float*)d_in[0];   // (4, 256, 256)
    const float* y    = (const float*)d_in[1];   // (4, 64, 256)
    const float* W    = (const float*)d_in[2];   // (1024, 256)
    const float* bias = (const float*)d_in[3];   // (1024,)
    float* out = (float*)d_out;                  // (4, 256, 64, 1024)

    dim3 grid(VOCAB_ / 256, T_, B_);
    dim3 block(256, 1, 1);

    if (ws_size >= WS_NEED_) {
        _Float16* ws  = (_Float16*)d_ws;
        _Float16* Wh  = ws + WH_OFF_;
        _Float16* xh  = ws + XH_OFF_;
        _Float16* yh  = ws + YH_OFF_;
        cvt_f16_kernel<<<dim3(WN_ / (8 * 256)), block, 0, stream>>>(x, y, W, xh, yh, Wh);
        joint_wmma_f16_kernel<<<grid, block, 0, stream>>>(xh, yh, Wh, bias, out);
    } else {
        joint_wmma_f32src_kernel<<<grid, block, 0, stream>>>(x, y, W, bias, out);
    }
}
